// Linear_83923660963882
// MI455X (gfx1250) — 8-bit matrix operand order repaired and checked by reading, compiled, not run on hardware
//
#include <hip/hip_runtime.h>

typedef int v8i __attribute__((ext_vector_type(8)));

#define MDIM 4096
#define NDIM 4096
#define KDIM 4096
#define KT   (KDIM / 64)

__global__ __launch_bounds__(256) void pack_a_kernel(const int* __restrict__ a,
                                                     unsigned* __restrict__ pA) {
  int i    = blockIdx.x * 256 + threadIdx.x;
  int tile = i >> 8;
  int q    = i & 255;
  int l    = q >> 3;
  int v    = q & 7;
  int mt   = tile / KT;
  int kt   = tile % KT;
  int m    = mt * 16 + (l & 15);
  int kb   = kt * 64 + ((v >> 2) << 5) + ((l >> 4) << 4) + ((v & 3) << 2);
  const int* s = a + (size_t)m * KDIM + kb;
  unsigned d = (unsigned)(s[0] & 255)
             | ((unsigned)(s[1] & 255) << 8)
             | ((unsigned)(s[2] & 255) << 16)
             | ((unsigned)(s[3] & 255) << 24);
  pA[i] = d;
}

__global__ __launch_bounds__(256) void pack_w_kernel(const int* __restrict__ w,
                                                     unsigned* __restrict__ pW) {
  __shared__ unsigned char lds[64 * 64];
  const int t  = threadIdx.x;
  const int kt = blockIdx.x;
  const int nb = blockIdx.y;

  const int col  = t & 63;
  const int row4 = t >> 6;
#pragma unroll
  for (int it = 0; it < 16; ++it) {
    int kl  = it * 4 + row4;
    int val = w[(size_t)(kt * 64 + kl) * NDIM + nb * 64 + col];
    lds[kl * 64 + col] = (unsigned char)(val & 255);
  }
  __syncthreads();

  const int j     = t >> 6;
  const int qb    = (t & 63) * 4;
  const int ntile = nb * 4 + j;
  unsigned* dst = pW + ((size_t)ntile * KT + kt) * 256 + qb;
#pragma unroll
  for (int d = 0; d < 4; ++d) {
    int q   = qb + d;
    int l   = q >> 3;
    int v   = q & 7;
    int nl  = j * 16 + (l & 15);
    int kbl = ((v >> 2) << 5) + ((l >> 4) << 4) + ((v & 3) << 2);
    unsigned dw = (unsigned)lds[kbl * 64 + nl]
                | ((unsigned)lds[(kbl + 1) * 64 + nl] << 8)
                | ((unsigned)lds[(kbl + 2) * 64 + nl] << 16)
                | ((unsigned)lds[(kbl + 3) * 64 + nl] << 24);
    dst[d] = dw;
  }
}

__global__ __launch_bounds__(256) void fragsum_kernel(const unsigned* __restrict__ p,
                                                      int* __restrict__ sums) {
  __shared__ int sm[16];
  const int t = threadIdx.x;
  if (t < 16) sm[t] = 0;
  __syncthreads();

  const int tile = blockIdx.x;
  const unsigned* base = p + (size_t)tile * KT * 256;
  int partial = 0;
  for (int kt = 0; kt < KT; ++kt) {
    unsigned d = base[kt * 256 + t];
    partial += (int)(d & 255) + (int)((d >> 8) & 255) +
               (int)((d >> 16) & 255) + (int)(d >> 24);
  }
  atomicAdd(&sm[(t >> 3) & 15], partial);
  __syncthreads();
  if (t < 16) sums[tile * 16 + t] = sm[t];
}

__global__ __launch_bounds__(512) void gemm_iu8_kernel(
    const unsigned* __restrict__ pA, const unsigned* __restrict__ pW,
    const int* __restrict__ rs, const int* __restrict__ cs,
    const float* __restrict__ a_s, const float* __restrict__ a_o,
    const float* __restrict__ w_s, const float* __restrict__ w_o,
    float* __restrict__ out) {
  const int lane = threadIdx.x & 31;
  const int wave = threadIdx.x >> 5;
  const int wr   = wave & 3;
  const int wc   = wave >> 2;

  const int mt0 = blockIdx.y * 16 + wr * 4;
  const int nt0 = blockIdx.x * 8 + wc * 2;

  v8i acc[4][2];
#pragma unroll
  for (int i = 0; i < 4; ++i)
#pragma unroll
    for (int j = 0; j < 2; ++j) acc[i][j] = (v8i){0, 0, 0, 0, 0, 0, 0, 0};

  const unsigned* aBase = pA + (size_t)mt0 * KT * 256 + lane * 8;
  const unsigned* bBase = pW + (size_t)nt0 * KT * 256 + lane * 8;

  const unsigned* pfA = aBase + 256;
  const unsigned* pfB = bBase + 256;

  for (int kt = 0; kt < KT; ++kt) {
    asm volatile("global_prefetch_b8 %0, off\n\t"
                 "global_prefetch_b8 %0, off offset:65536\n\t"
                 "global_prefetch_b8 %0, off offset:131072\n\t"
                 "global_prefetch_b8 %0, off offset:196608\n\t"
                 "v_add_nc_u64_e32 %0, 0x400, %0"
                 : "+v"(pfA));
    asm volatile("global_prefetch_b8 %0, off\n\t"
                 "global_prefetch_b8 %0, off offset:65536\n\t"
                 "v_add_nc_u64_e32 %0, 0x400, %0"
                 : "+v"(pfB));

    const unsigned* aCur = aBase + (size_t)kt * 256;
    const unsigned* bCur = bBase + (size_t)kt * 256;

    v8i af[4], bf[2];
#pragma unroll
    for (int i = 0; i < 4; ++i)
      af[i] = *(const v8i*)(aCur + (size_t)i * KT * 256);
#pragma unroll
    for (int j = 0; j < 2; ++j)
      bf[j] = *(const v8i*)(bCur + (size_t)j * KT * 256);

#pragma unroll
    for (int i = 0; i < 4; ++i)
#pragma unroll
      for (int j = 0; j < 2; ++j)
        acc[i][j] = __builtin_amdgcn_wmma_i32_16x16x64_iu8(
             false, af[i],  false, bf[j], acc[i][j],
             false,  false);
  }

  const float sA = a_s[0], oA = a_o[0], sW = w_s[0], oW = w_o[0];
  const float s_ab  = sA * sW;
  const float s_row = sA * oW;
  const float s_col = oA * sW;
  const float c0    = (float)KDIM * oA * oW;

  const int halfOff = (lane >> 4) << 3;

#pragma unroll
  for (int i = 0; i < 4; ++i) {
    const int mrow0 = (mt0 + i) * 16;
#pragma unroll
    for (int j = 0; j < 2; ++j) {
      const int ncol   = (nt0 + j) * 16 + (lane & 15);
      const float cadd = (float)cs[ncol] * s_col + c0;
#pragma unroll
      for (int r = 0; r < 8; ++r) {
        const int m = mrow0 + r + halfOff;
        out[(size_t)m * NDIM + ncol] =
            (float)acc[i][j][r] * s_ab + (float)rs[m] * s_row + cadd;
      }
    }
  }
}

extern "C" void kernel_launch(void* const* d_in, const int* in_sizes, int n_in,
                              void* d_out, int out_size, void* d_ws, size_t ws_size,
                              hipStream_t stream) {
  const int*   a   = (const int*)d_in[0];
  const float* a_s = (const float*)d_in[1];
  const float* a_o = (const float*)d_in[2];
  const int*   w   = (const int*)d_in[3];
  const float* w_s = (const float*)d_in[4];
  const float* w_o = (const float*)d_in[5];
  float*       out = (float*)d_out;

  unsigned char* ws = (unsigned char*)d_ws;
  unsigned* pA = (unsigned*)ws;
  unsigned* pW = (unsigned*)(ws + (size_t)MDIM * KDIM);
  int*      rs = (int*)(ws + 2ull * MDIM * KDIM);
  int*      cs = rs + MDIM;

  const int packBlocks = (MDIM * (KDIM / 4)) / 256;
  pack_a_kernel<<<packBlocks, 256, 0, stream>>>(a, pA);

  dim3 wgrid(KT, NDIM / 64);
  pack_w_kernel<<<wgrid, 256, 0, stream>>>(w, pW);

  fragsum_kernel<<<MDIM / 16, 256, 0, stream>>>(pA, rs);
  fragsum_kernel<<<NDIM / 16, 256, 0, stream>>>(pW, cs);

  dim3 grid(NDIM / 128, MDIM / 256);
  gemm_iu8_kernel<<<grid, 512, 0, stream>>>(pA, pW, rs, cs, a_s, a_o, w_s, w_o, out);
}
